// S4D_14139032339102
// MI455X (gfx1250) — matrix operand order repaired and checked by reading, not compiled, not run on hardware
//
#include <hip/hip_runtime.h>
#include <hip/hip_bf16.h>

// ---------------- problem constants ----------------
#define Bb   8
#define Hh   512
#define Nn   32            // N/2 modes
#define Ll   4096
#define KPAD 320
#define KLEN (Ll + KPAD)   // reversed-k row length (zero tail)
#define KWIN 288           // staged window: 256 block columns + 31 + pad

typedef __attribute__((ext_vector_type(16))) _Float16 v16h;
typedef __attribute__((ext_vector_type(8)))  float    v8f;

union V16H { uint4 q[2]; v16h v; };

// ---------------- TDM support (probe-gated) ----------------
#if __has_builtin(__builtin_amdgcn_tensor_load_to_lds) && \
    __has_builtin(__builtin_amdgcn_s_wait_tensorcnt)
#define USE_TDM 1
typedef __attribute__((ext_vector_type(4))) unsigned int u32x4;
typedef __attribute__((ext_vector_type(8))) int          i32x8;
typedef __attribute__((ext_vector_type(4))) int          i32x4;

// Build a D# for a (up to) 2D tile of 2-byte elements and issue the DMA.
// group0: count=1 | lds_addr | global_addr(57b) | type=2
// group1: data_size=2B | tensor_dim0/1 | tile_dim0/1 | dim0 stride
__device__ __forceinline__ void tdm_load_2d(unsigned lds_off, const void* gptr,
                                            unsigned tdim0, unsigned tdim1,
                                            unsigned long long stride0_elems,
                                            unsigned tile0, unsigned tile1) {
    unsigned long long ga = (unsigned long long)(uintptr_t)gptr;
    u32x4 g0;
    g0.x = 1u;                                   // count = 1 valid descriptor
    g0.y = lds_off;                              // LDS byte address
    g0.z = (unsigned)ga;                         // global addr [31:0]
    g0.w = (unsigned)(ga >> 32) | 0x80000000u;   // addr[56:32] | type=2<<30
    i32x8 g1;
    g1[0] = 0x00010000;                          // data_size = 1 (2 bytes)
    g1[1] = (int)((tdim0 & 0xffffu) << 16);                      // tensor_dim0 lo
    g1[2] = (int)((tdim0 >> 16) | ((tdim1 & 0xffffu) << 16));    // dim0 hi | dim1 lo
    g1[3] = (int)((tdim1 >> 16) | (tile0 << 16));                // dim1 hi | tile_dim0
    g1[4] = (int)(tile1 & 0xffffu);                              // tile_dim1 (0=1 row)
    g1[5] = (int)(stride0_elems & 0xffffffffu);                  // dim0 stride lo
    g1[6] = (int)((stride0_elems >> 32) & 0xffffu);              // dim0 stride hi
    g1[7] = 0;
#if __clang_major__ >= 23
    __builtin_amdgcn_tensor_load_to_lds(g0, g1, (i32x4)0, (i32x4)0, (i32x8)0, 0);
#else
    __builtin_amdgcn_tensor_load_to_lds(g0, g1, (i32x4)0, (i32x4)0, 0);
#endif
}
#endif

// =====================================================================
// Kernel 1: build k[h,l] = 2*Re( sum_n Ck[h,n] * exp(dtA[h,n]*l) )
// Stored reversed + zero-padded:  krev[h][i]  = k[h, L-1-i]  (0 for i>=L)
//                                 krev2[h][i] = krev[h][i+1]   (parity copy)
// =====================================================================
__global__ void s4d_kernelgen(const float* __restrict__ log_dt,
                              const float* __restrict__ C_re,
                              const float* __restrict__ C_im,
                              const float* __restrict__ log_A_real,
                              const float* __restrict__ A_imag,
                              _Float16* __restrict__ krev,
                              _Float16* __restrict__ krev2) {
    const int h = blockIdx.x;
    const int t = threadIdx.x;
    const float dt = expf(log_dt[h]);

    __shared__ float wr256[Nn], wi256[Nn];
    if (t < Nn) {
        float Are = -expf(log_A_real[h * Nn + t]);
        float Aim = A_imag[h * Nn + t];
        float ar = Are * dt, ai = Aim * dt;
        float e = expf(ar * 256.f);
        wr256[t] = e * cosf(ai * 256.f);
        wi256[t] = e * sinf(ai * 256.f);
    }
    __syncthreads();

    float zr[Nn], zi[Nn], Ckr[Nn], Cki[Nn];
#pragma unroll 4
    for (int n = 0; n < Nn; ++n) {
        float Are = -expf(log_A_real[h * Nn + n]);
        float Aim = A_imag[h * Nn + n];
        float ar = Are * dt, ai = Aim * dt;
        float e = expf(ar * (float)t);
        zr[n] = e * cosf(ai * (float)t);
        zi[n] = e * sinf(ai * (float)t);
        float ew = expf(ar);
        float wr = ew * cosf(ai), wi = ew * sinf(ai);
        float Er = wr - 1.f, Ei = wi;
        float inv = 1.f / (Are * Are + Aim * Aim);
        float Dr = (Er * Are + Ei * Aim) * inv;
        float Di = (Ei * Are - Er * Aim) * inv;
        float cr = C_re[h * Nn + n], ci = C_im[h * Nn + n];
        Ckr[n] = cr * Dr - ci * Di;
        Cki[n] = cr * Di + ci * Dr;
    }

    _Float16* kr  = krev  + (size_t)h * KLEN;
    _Float16* kr2 = krev2 + (size_t)h * KLEN;
#pragma unroll 1
    for (int it = 0; it < 16; ++it) {
        const int l = t + 256 * it;
        float acc = 0.f;
#pragma unroll
        for (int n = 0; n < Nn; ++n) acc += Ckr[n] * zr[n] - Cki[n] * zi[n];
        const float kv = 2.f * acc;
        kr[Ll - 1 - l] = (_Float16)kv;
        if (l < Ll - 1) kr2[Ll - 2 - l] = (_Float16)kv;
#pragma unroll
        for (int n = 0; n < Nn; ++n) {
            float a = zr[n], b = zi[n];
            float cr = wr256[n], ci = wi256[n];
            zr[n] = a * cr - b * ci;
            zi[n] = a * ci + b * cr;
        }
    }
    for (int i = t; i < KPAD; i += 256)      kr [Ll + i]     = (_Float16)0.f;
    for (int i = t; i < KPAD + 1; i += 256)  kr2[Ll - 1 + i] = (_Float16)0.f;
}

// =====================================================================
// Kernel 2: f32 -> f16 conversion
// =====================================================================
__global__ void s4d_cvt_f16(const float* __restrict__ src,
                            _Float16* __restrict__ dst, int n) {
    int i = blockIdx.x * blockDim.x + threadIdx.x;
    if (i < n) dst[i] = (_Float16)src[i];
}

// =====================================================================
// Kernel 3: causal Toeplitz convolution via WMMA, TDM double-buffered.
// Block = 8 waves on one h covering 256 l-columns; each wave owns TWO
// adjacent 16-wide l-tiles.  Per 32-wide t-chunk, TDM stages into LDS:
//   kwin/kwin2 [288] : reversed-k window (+ parity copy)
//   awin       [256] : u[b=0..7][t0..t0+32) tile (A-matrix source)
// Chunk c+1 is DMA'd into the ping-pong buffer while chunk c computes
// (TDM is in-order per wave -> s_wait_tensorcnt(3) guards buffer c).
// One contiguous run of 16 ds_load_b32 feeds BOTH WMMAs of the wave.
// Epilogue: + u*D, erf-GELU, store y^T (b,l,h) f16 for the GLU GEMM.
// =====================================================================
__global__ __launch_bounds__(256)
void s4d_conv_wmma(const _Float16* __restrict__ u16,
                   const _Float16* __restrict__ krev,
                   const _Float16* __restrict__ krev2,
                   const float*    __restrict__ uf32,
                   const float*    __restrict__ Dvec,
                   _Float16* __restrict__ y_t) {
    __shared__ alignas(16) _Float16 kwin [2][KWIN];
    __shared__ alignas(16) _Float16 kwin2[2][KWIN];
    __shared__ alignas(16) _Float16 awin [2][256];

    const int lane = threadIdx.x & 31;
    const int w    = threadIdx.x >> 5;          // wave in block: 0..7
    const int h    = blockIdx.x >> 4;           // 512 h
    const int l0b  = (blockIdx.x & 15) << 8;    // block's first l column
    const int l0   = l0b + (w << 5);            // wave's low l-tile
    const int hs   = lane >> 4;
    const int lm   = lane & 15;

    v8f acc0 = {0.f,0.f,0.f,0.f,0.f,0.f,0.f,0.f};   // tile l0
    v8f acc1 = {0.f,0.f,0.f,0.f,0.f,0.f,0.f,0.f};   // tile l0+16

    // block-wide chunk count (extra chunks for low-l0 waves hit the zero tail)
    const int nch = ((l0b + 255) >> 5) + 1;

    const _Float16* krA = krev  + (size_t)h * KLEN;
    const _Float16* krB = krev2 + (size_t)h * KLEN;
    const _Float16* uBase = u16 + (size_t)h * Ll;

    // chunk-invariant Toeplitz offset of the HIGH tile inside the window;
    // low tile's 16 halves follow contiguously (rel_lo = rel_hi + 16).
    const int relh = 255 - (w << 5) - 16 - lm + (hs << 4);   // 0..255
    const uint* kpb[2];
#pragma unroll
    for (int p = 0; p < 2; ++p)
        kpb[p] = (relh & 1) ? ((const uint*)&kwin2[p][0] + ((relh - 1) >> 1))
                            : ((const uint*)&kwin [p][0] + (relh >> 1));

#ifdef USE_TDM
    // preload chunk 0 into buffer 0
    if (threadIdx.x < 32) {
        const int i0 = Ll - 256 - l0b;
        tdm_load_2d((unsigned)(uintptr_t)&kwin [0][0], krA + i0,
                    KWIN, 1u, (unsigned long long)KWIN, KWIN, 0u);
        tdm_load_2d((unsigned)(uintptr_t)&kwin2[0][0], krB + i0,
                    KWIN, 1u, (unsigned long long)KWIN, KWIN, 0u);
        tdm_load_2d((unsigned)(uintptr_t)&awin [0][0], uBase,
                    32u, 8u, (unsigned long long)Hh * Ll, 32u, 8u);
    }
#endif

    for (int c = 0; c < nch; ++c) {
        const int buf = c & 1;
#ifdef USE_TDM
        if (threadIdx.x < 32) {
            if (c + 1 < nch) {   // stream chunk c+1 into the other buffer
                const int nt0   = (c + 1) << 5;
                const int i_min = Ll - 256 - l0b + nt0;
                const int nb    = buf ^ 1;
                tdm_load_2d((unsigned)(uintptr_t)&kwin [nb][0], krA + i_min,
                            KWIN, 1u, (unsigned long long)KWIN, KWIN, 0u);
                tdm_load_2d((unsigned)(uintptr_t)&kwin2[nb][0], krB + i_min,
                            KWIN, 1u, (unsigned long long)KWIN, KWIN, 0u);
                tdm_load_2d((unsigned)(uintptr_t)&awin [nb][0], uBase + nt0,
                            32u, 8u, (unsigned long long)Hh * Ll, 32u, 8u);
                __builtin_amdgcn_s_wait_tensorcnt(3);  // chunk c's 3 loads done
            } else {
                __builtin_amdgcn_s_wait_tensorcnt(0);
            }
        }
#else
        {   // synchronous fallback staging of chunk c
            const int t0    = c << 5;
            const int i_min = Ll - 256 - l0b + t0;
            int j = threadIdx.x;
            if (j < KWIN + 1) {
                _Float16 v = krA[i_min + j];
                if (j < KWIN) kwin[buf][j] = v;
                if (j > 0)    kwin2[buf][j - 1] = v;
            }
            int bb = threadIdx.x >> 5, tt = threadIdx.x & 31;
            awin[buf][threadIdx.x] = u16[((size_t)bb * Hh + h) * Ll + t0 + tt];
        }
#endif
        __syncthreads();   // buffer `buf` ready for all waves

        // ---- A tile: u rows b=0..7, rows 8..15 zero (shared by both WMMAs)
        V16H A;
        if (lm < 8) {
            const uint4* ap = (const uint4*)(&awin[buf][0] + lm * 32 + (hs << 4));
            A.q[0] = ap[0]; A.q[1] = ap[1];
        } else {
            A.q[0] = make_uint4(0u, 0u, 0u, 0u);
            A.q[1] = make_uint4(0u, 0u, 0u, 0u);
        }
        // ---- B tiles: one contiguous 16-dword LDS run covers both tiles
        const uint* kq = kpb[buf];
        V16H B1, B0;
        B1.q[0] = make_uint4(kq[0],  kq[1],  kq[2],  kq[3]);   // tile l0+16
        B1.q[1] = make_uint4(kq[4],  kq[5],  kq[6],  kq[7]);
        B0.q[0] = make_uint4(kq[8],  kq[9],  kq[10], kq[11]);  // tile l0
        B0.q[1] = make_uint4(kq[12], kq[13], kq[14], kq[15]);

        acc0 = __builtin_amdgcn_wmma_f32_16x16x32_f16(
                   false, A.v, false, B0.v, (short)0, acc0, false, false);
        acc1 = __builtin_amdgcn_wmma_f32_16x16x32_f16(
                   false, A.v, false, B1.v, (short)0, acc1, false, false);

        __syncthreads();   // all reads of `buf` done before it is re-filled
    }

    // ---- epilogue: lanes 0..15 hold all 8 valid batch rows (M = 0..7)
    if (hs == 0) {
        const int l = l0 + lm;
        const float Dh = Dvec[h];
#pragma unroll
        for (int r = 0; r < 8; ++r) {
            const int b = r;
            float v0 = acc0[r] + uf32[((size_t)b * Hh + h) * Ll + l] * Dh;
            float g0 = 0.5f * v0 * (1.f + erff(v0 * 0.70710678118654752f));
            y_t[((size_t)(b * Ll + l)) * Hh + h] = (_Float16)g0;
            float v1 = acc1[r] + uf32[((size_t)b * Hh + h) * Ll + l + 16] * Dh;
            float g1 = 0.5f * v1 * (1.f + erff(v1 * 0.70710678118654752f));
            y_t[((size_t)(b * Ll + l + 16)) * Hh + h] = (_Float16)g1;
        }
    }
}

// =====================================================================
// Kernel 4: GLU output projection.  z = W @ y + b ; out = a * sigmoid(g)
// Each wave: one 16-row (a,g) pair x TWO 16-wide column tiles -> A tiles
// loaded once per K-chunk feed 4 WMMAs.
// =====================================================================
__global__ __launch_bounds__(256)
void s4d_glu_wmma(const _Float16* __restrict__ Wf16,
                  const _Float16* __restrict__ y_t,
                  const float*    __restrict__ bias,
                  float* __restrict__ out) {
    const int lane = threadIdx.x & 31;
    const int wid  = blockIdx.x * (blockDim.x >> 5) + (threadIdx.x >> 5);
    const int ot   = wid >> 10;               // 32 o-pair tiles
    const int ct   = wid & 1023;              // 1024 column-pair tiles
    const int o0   = ot << 4;                 // a-rows [o0, o0+16)
    const int b    = ct >> 7;
    const int l0   = (ct & 127) << 5;         // 32 columns per wave
    const int lm   = lane & 15;
    const int hs   = lane >> 4;

    v8f accA0 = {0.f,0.f,0.f,0.f,0.f,0.f,0.f,0.f};
    v8f accG0 = {0.f,0.f,0.f,0.f,0.f,0.f,0.f,0.f};
    v8f accA1 = {0.f,0.f,0.f,0.f,0.f,0.f,0.f,0.f};
    v8f accG1 = {0.f,0.f,0.f,0.f,0.f,0.f,0.f,0.f};

    const _Float16* wRowA = Wf16 + (size_t)(o0 + lm) * Hh;
    const _Float16* wRowG = wRowA + (size_t)512 * Hh;
    const _Float16* yCol0 = y_t + ((size_t)(b * Ll) + l0 + lm) * Hh;
    const _Float16* yCol1 = yCol0 + (size_t)16 * Hh;

#pragma unroll 2
    for (int k0 = 0; k0 < Hh; k0 += 32) {
        V16H A1, A2, B0, B1;
        const uint4* pa = (const uint4*)(wRowA + k0 + (hs << 4));
        A1.q[0] = pa[0]; A1.q[1] = pa[1];
        const uint4* pg = (const uint4*)(wRowG + k0 + (hs << 4));
        A2.q[0] = pg[0]; A2.q[1] = pg[1];
        const uint4* pb0 = (const uint4*)(yCol0 + k0 + (hs << 4));
        B0.q[0] = pb0[0]; B0.q[1] = pb0[1];
        const uint4* pb1 = (const uint4*)(yCol1 + k0 + (hs << 4));
        B1.q[0] = pb1[0]; B1.q[1] = pb1[1];
        __builtin_prefetch(yCol0 + k0 + 32, 0, 1);

        accA0 = __builtin_amdgcn_wmma_f32_16x16x32_f16(
                    false, A1.v, false, B0.v, (short)0, accA0, false, false);
        accG0 = __builtin_amdgcn_wmma_f32_16x16x32_f16(
                    false, A2.v, false, B0.v, (short)0, accG0, false, false);
        accA1 = __builtin_amdgcn_wmma_f32_16x16x32_f16(
                    false, A1.v, false, B1.v, (short)0, accA1, false, false);
        accG1 = __builtin_amdgcn_wmma_f32_16x16x32_f16(
                    false, A2.v, false, B1.v, (short)0, accG1, false, false);
    }

    const int l = l0 + lm;
#pragma unroll
    for (int r = 0; r < 8; ++r) {
        const int m = (hs << 3) + r;          // D-tile M index
        const int o = o0 + m;
        const float ba = bias[o];
        const float bg = bias[o + 512];
        float za0 = accA0[r] + ba, zg0 = accG0[r] + bg;
        out[((size_t)b * Hh + o) * Ll + l]      = za0 / (1.f + expf(-zg0));
        float za1 = accA1[r] + ba, zg1 = accG1[r] + bg;
        out[((size_t)b * Hh + o) * Ll + l + 16] = za1 / (1.f + expf(-zg1));
    }
}

// =====================================================================
extern "C" void kernel_launch(void* const* d_in, const int* in_sizes, int n_in,
                              void* d_out, int out_size, void* d_ws, size_t ws_size,
                              hipStream_t stream) {
    const float* u          = (const float*)d_in[0];
    const float* log_dt     = (const float*)d_in[1];
    const float* C_re       = (const float*)d_in[2];
    const float* C_im       = (const float*)d_in[3];
    const float* log_A_real = (const float*)d_in[4];
    const float* A_imag     = (const float*)d_in[5];
    const float* Dvec       = (const float*)d_in[6];
    const float* W          = (const float*)d_in[7];
    const float* bias       = (const float*)d_in[8];
    float* out = (float*)d_out;

    char* ws = (char*)d_ws;
    size_t off = 0;
    auto carve = [&](size_t bytes) -> void* {
        void* p = ws + off;
        off = (off + bytes + 255) & ~(size_t)255;
        return p;
    };
    const size_t BHL = (size_t)Bb * Hh * Ll;
    _Float16* u16   = (_Float16*)carve(BHL * 2);                  // 32 MB
    _Float16* krev  = (_Float16*)carve((size_t)Hh * KLEN * 2);    // 4.5 MB
    _Float16* krev2 = (_Float16*)carve((size_t)Hh * KLEN * 2);    // 4.5 MB
    _Float16* W16   = (_Float16*)carve((size_t)1024 * Hh * 2);    // 1 MB
    _Float16* y_t   = (_Float16*)carve(BHL * 2);                  // 32 MB
    (void)ws_size; (void)n_in; (void)in_sizes; (void)out_size;

    s4d_cvt_f16<<<(int)((BHL + 255) / 256), 256, 0, stream>>>(u, u16, (int)BHL);
    s4d_cvt_f16<<<(1024 * Hh + 255) / 256, 256, 0, stream>>>(W, W16, 1024 * Hh);

    s4d_kernelgen<<<Hh, 256, 0, stream>>>(log_dt, C_re, C_im, log_A_real,
                                          A_imag, krev, krev2);

    // 512 h * 16 column-blocks (256 cols each) = 8192 blocks of 8 waves
    s4d_conv_wmma<<<8192, 256, 0, stream>>>(u16, krev, krev2, u, Dvec, y_t);

    // 32 o-pair tiles * 1024 column-pair tiles = 32768 waves -> 4096 blocks
    s4d_glu_wmma<<<4096, 256, 0, stream>>>(W16, y_t, bias, out);
}
